// Model_85255100825951
// MI455X (gfx1250) — compile-verified, matrix operand order checked by reading, not run on hardware
//
#include <hip/hip_runtime.h>
#include <hip/hip_bf16.h>
#include <math.h>

typedef __attribute__((ext_vector_type(16))) __bf16 v16bf;
typedef __attribute__((ext_vector_type(8)))  float  v8f;

#define Tn   2048
#define Dm   1024
#define En   8
#define Fh   2048
#define VOC  32000
#define NHOP 2
#define CAP  640        // int(1.25 * 2 * 2048 / 8)

#define BM 128
#define BN 128

// Detect gfx1250 async global->LDS support (ASYNCcnt path).
#if defined(__has_builtin)
#  if __has_builtin(__builtin_amdgcn_global_load_async_to_lds_b128)
#    define ASYNC_LDS 1
#  endif
#endif

#ifdef ASYNC_LDS
// double-buffered async pipeline: K-step 32, 96B LDS rows (32B-aligned frags)
#  define BK   32
#  define LDSK 48
#else
// synchronous fallback: K-step 64, 160B LDS rows
#  define BK   64
#  define LDSK 80
#endif

// ---------------------------------------------------------------- helpers
__device__ __forceinline__ float gelu_tanh(float x) {
    // tanh-approx GELU, branch-free via hardware exp: tanh(u) = 1 - 2/(1+e^{2u})
    float u = 0.7978845608028654f * (x + 0.044715f * x * x * x);
    float e = __expf(2.0f * u);
    float t = 1.0f - 2.0f / (e + 1.0f);
    return 0.5f * x * (1.0f + t);
}

__device__ __forceinline__ v8f wmma_bf16(v16bf a, v16bf b, v8f c) {
    return __builtin_amdgcn_wmma_f32_16x16x32_bf16(
        /*neg_a=*/false, a, /*neg_b=*/false, b,
        /*c_mod=*/(short)0, c, /*reuse_a=*/false, /*reuse_b=*/false);
}

#ifdef ASYNC_LDS
typedef int v4i_vec __attribute__((vector_size(16)));
typedef __attribute__((address_space(1))) v4i_vec* as1_v4i;
typedef __attribute__((address_space(3))) v4i_vec* as3_v4i;

__device__ __forceinline__ void async_copy_b128(const __bf16* g, __bf16* l) {
    __builtin_amdgcn_global_load_async_to_lds_b128(
        (as1_v4i)(__bf16*)g, (as3_v4i)l, /*offset=*/0, /*cpol=*/0);
}
__device__ __forceinline__ void wait_async_all() {
#  if __has_builtin(__builtin_amdgcn_s_wait_asynccnt)
    __builtin_amdgcn_s_wait_asynccnt(0);
#  else
    asm volatile("s_wait_asynccnt 0x0" ::: "memory");
#  endif
}
__device__ __forceinline__ void wait_async_4() {
#  if __has_builtin(__builtin_amdgcn_s_wait_asynccnt)
    __builtin_amdgcn_s_wait_asynccnt(4);
#  else
    asm volatile("s_wait_asynccnt 0x4" ::: "memory");
#  endif
}
#endif

// ------------------------------------------------ one-time bf16 conversion
__global__ __launch_bounds__(256)
void conv_bf16(const float* __restrict__ in, __bf16* __restrict__ out) {
    size_t i = ((size_t)blockIdx.x * 256 + threadIdx.x) * 4;
    float4 v = *(const float4*)(in + i);
    out[i + 0] = (__bf16)v.x;
    out[i + 1] = (__bf16)v.y;
    out[i + 2] = (__bf16)v.z;
    out[i + 3] = (__bf16)v.w;
}

// transpose + convert: in f32 [R,C] -> out bf16 [C,R]; grid (C/32, R/32, nmat)
__global__ __launch_bounds__(256)
void conv_transpose_bf16(const float* __restrict__ in, __bf16* __restrict__ out,
                         int R, int C) {
    __shared__ float tile[32][33];
    const float* src = in + (size_t)blockIdx.z * R * C;
    __bf16*      dst = out + (size_t)blockIdx.z * R * C;
    int c0 = blockIdx.x * 32, r0 = blockIdx.y * 32;
    int x = threadIdx.x & 31, ys = threadIdx.x >> 5;
    #pragma unroll
    for (int y = ys; y < 32; y += 8)
        tile[y][x] = src[(size_t)(r0 + y) * C + c0 + x];
    __syncthreads();
    #pragma unroll
    for (int y = ys; y < 32; y += 8)
        dst[(size_t)(c0 + y) * R + r0 + x] = (__bf16)tile[x][y];
}

// ---------------------------------------------------------------- gather
__global__ __launch_bounds__(256)
void moe_gather(const int* __restrict__ ids, const float* __restrict__ emb,
                float* __restrict__ h, __bf16* __restrict__ hbf) {
    int t = blockIdx.x;
    const float* src = emb + (size_t)ids[t] * Dm;
    for (int d = threadIdx.x; d < Dm; d += 256) {
        float v = src[d];
        h[(size_t)t * Dm + d]   = v;
        hbf[(size_t)t * Dm + d] = (__bf16)v;
    }
}

// ---------------------------------------------------------------- router
__global__ __launch_bounds__(256)
void moe_router(const float* __restrict__ h, const float* __restrict__ rw,
                float* __restrict__ probs, unsigned* __restrict__ maskbits) {
    int t = blockIdx.x * blockDim.x + threadIdx.x;
    if (t >= Tn) return;
    float acc[En];
    #pragma unroll
    for (int e = 0; e < En; ++e) acc[e] = 0.f;
    const float* hp = h + (size_t)t * Dm;
    for (int d = 0; d < Dm; ++d) {
        float hv = hp[d];
        const float* r = rw + (size_t)d * En;
        #pragma unroll
        for (int e = 0; e < En; ++e) acc[e] = fmaf(hv, r[e], acc[e]);
    }
    float m = acc[0];
    #pragma unroll
    for (int e = 1; e < En; ++e) m = fmaxf(m, acc[e]);
    float s = 0.f;
    float p[En];
    #pragma unroll
    for (int e = 0; e < En; ++e) { p[e] = __expf(acc[e] - m); s += p[e]; }
    float inv = 1.0f / s;
    #pragma unroll
    for (int e = 0; e < En; ++e) { p[e] *= inv; probs[(size_t)t * En + e] = p[e]; }
    int i1 = 0;
    #pragma unroll
    for (int e = 1; e < En; ++e) if (p[e] > p[i1]) i1 = e;
    int i2 = (i1 == 0) ? 1 : 0;
    #pragma unroll
    for (int e = 0; e < En; ++e) if (e != i1 && p[e] > p[i2]) i2 = e;
    maskbits[t] = (1u << i1) | (1u << i2);
}

// ------------------------------------------------- capacity (serial scan)
__global__ void moe_capacity(const float* __restrict__ probs,
                             const unsigned* __restrict__ maskbits,
                             float* __restrict__ kept) {
    int e = threadIdx.x;
    if (e >= En) return;
    int cnt = 0;
    for (int t = 0; t < Tn; ++t) {
        float kw = 0.f;
        if ((maskbits[t] >> e) & 1u) {
            ++cnt;
            if (cnt <= CAP) kw = probs[(size_t)t * En + e];
        }
        kept[(size_t)t * En + e] = kw;
    }
}

// -------------------------------------------------------- unified GEMM core
// A: [M,KD] bf16 K-contiguous.  B: [N,KD] bf16 K-contiguous.
template <int KD>
__device__ __forceinline__ void gemm_core_bf16(const __bf16* __restrict__ A,
                                               const __bf16* __restrict__ B,
                                               int m0, int n0, v8f (&acc)[4][2]) {
    const int tid  = threadIdx.x;
    const int lane = tid & 31, w = tid >> 5;
    const int wrow = w >> 2, wcol = w & 3;
    const int l15 = lane & 15, khb = (lane >> 4) * 16;

    #pragma unroll
    for (int i = 0; i < 4; ++i)
        #pragma unroll
        for (int j = 0; j < 2; ++j)
            #pragma unroll
            for (int q = 0; q < 8; ++q) acc[i][j][q] = 0.f;

#ifdef ASYNC_LDS
    // ---- double-buffered async global->LDS pipeline (ASYNCcnt) ----
    __shared__ __align__(32) __bf16 As[2][BM][LDSK];
    __shared__ __align__(32) __bf16 Bs[2][BN][LDSK];
    const int row  = tid >> 1;             // 0..127 staging row
    const int koff = (tid & 1) * 16;       // halves within BK=32 (32B each)

    // issue tile 0 into buffer 0 (4 async b128 per thread)
    {
        const __bf16* ga = A + (size_t)(m0 + row) * KD + koff;
        const __bf16* gb = B + (size_t)(n0 + row) * KD + koff;
        async_copy_b128(ga,     &As[0][row][koff]);
        async_copy_b128(ga + 8, &As[0][row][koff + 8]);
        async_copy_b128(gb,     &Bs[0][row][koff]);
        async_copy_b128(gb + 8, &Bs[0][row][koff + 8]);
    }
    int buf = 0;
    for (int k0 = 0; k0 < KD; k0 += BK, buf ^= 1) {
        if (k0 + BK < KD) {  // issue next tile into the other buffer
            const __bf16* ga = A + (size_t)(m0 + row) * KD + k0 + BK + koff;
            const __bf16* gb = B + (size_t)(n0 + row) * KD + k0 + BK + koff;
            async_copy_b128(ga,     &As[buf ^ 1][row][koff]);
            async_copy_b128(ga + 8, &As[buf ^ 1][row][koff + 8]);
            async_copy_b128(gb,     &Bs[buf ^ 1][row][koff]);
            async_copy_b128(gb + 8, &Bs[buf ^ 1][row][koff + 8]);
            wait_async_4();              // current tile's 4 ops have landed
        } else {
            wait_async_all();
        }
        __syncthreads();                 // all threads' current tile in LDS
        v16bf af[4], bfm[2];
        #pragma unroll
        for (int i = 0; i < 4; ++i)
            af[i] = *(const v16bf*)&As[buf][wrow * 64 + i * 16 + l15][khb];
        #pragma unroll
        for (int j = 0; j < 2; ++j)
            bfm[j] = *(const v16bf*)&Bs[buf][wcol * 32 + j * 16 + l15][khb];
        #pragma unroll
        for (int i = 0; i < 4; ++i)
            #pragma unroll
            for (int j = 0; j < 2; ++j)
                acc[i][j] = wmma_bf16(af[i], bfm[j], acc[i][j]);
        __syncthreads();                 // protect buffer refilled next iter
    }
#else
    // ---- synchronous fallback: VGPR-staged, K-step 64 ----
    __shared__ __align__(32) __bf16 As[BM][LDSK];
    __shared__ __align__(32) __bf16 Bs[BN][LDSK];
    const int row  = tid >> 1;
    const int koff = (tid & 1) * 32;

    for (int k0 = 0; k0 < KD; k0 += BK) {
        const uint4* sa = (const uint4*)(A + (size_t)(m0 + row) * KD + k0 + koff);
        const uint4* sb = (const uint4*)(B + (size_t)(n0 + row) * KD + k0 + koff);
        uint4 a0 = sa[0], a1 = sa[1], a2 = sa[2], a3 = sa[3];
        uint4 b0 = sb[0], b1 = sb[1], b2 = sb[2], b3 = sb[3];
        if (k0 + BK < KD) {
            __builtin_prefetch(A + (size_t)(m0 + row) * KD + k0 + BK + koff, 0, 1);
            __builtin_prefetch(B + (size_t)(n0 + row) * KD + k0 + BK + koff, 0, 1);
        }
        uint4* da = (uint4*)&As[row][koff];
        da[0] = a0; da[1] = a1; da[2] = a2; da[3] = a3;
        uint4* db = (uint4*)&Bs[row][koff];
        db[0] = b0; db[1] = b1; db[2] = b2; db[3] = b3;
        __syncthreads();
        #pragma unroll
        for (int kk = 0; kk < BK; kk += 32) {
            v16bf af[4], bfm[2];
            #pragma unroll
            for (int i = 0; i < 4; ++i)
                af[i] = *(const v16bf*)&As[wrow * 64 + i * 16 + l15][kk + khb];
            #pragma unroll
            for (int j = 0; j < 2; ++j)
                bfm[j] = *(const v16bf*)&Bs[wcol * 32 + j * 16 + l15][kk + khb];
            #pragma unroll
            for (int i = 0; i < 4; ++i)
                #pragma unroll
                for (int j = 0; j < 2; ++j)
                    acc[i][j] = wmma_bf16(af[i], bfm[j], acc[i][j]);
        }
        __syncthreads();
    }
#endif
}

// -------------------------------- GEMM1: act = gelu(hbf @ w1T[e]^T) -> bf16
__global__ __launch_bounds__(256)
void moe_gemm1_gelu(const __bf16* __restrict__ A, const __bf16* __restrict__ B,
                    __bf16* __restrict__ Cact) {
    const int lane = threadIdx.x & 31, w = threadIdx.x >> 5;
    const int wrow = w >> 2, wcol = w & 3;
    const int m0 = blockIdx.y * BM, n0 = blockIdx.x * BN;
    v8f acc[4][2];
    gemm_core_bf16<Dm>(A, B, m0, n0, acc);
    const int mo = (lane >> 4) * 8, ncol = lane & 15;
    #pragma unroll
    for (int i = 0; i < 4; ++i)
        #pragma unroll
        for (int j = 0; j < 2; ++j) {
            int gn = n0 + wcol * 32 + j * 16 + ncol;
            #pragma unroll
            for (int r = 0; r < 8; ++r) {
                int gm = m0 + wrow * 64 + i * 16 + mo + r;
                Cact[(size_t)gm * Fh + gn] = (__bf16)gelu_tanh(acc[i][j][r]);
            }
        }
}

// ----- GEMM2: combined (+)= kept[t,e] * (act @ w2T[e]^T)  (f32 out)
__global__ __launch_bounds__(256)
void moe_gemm2_acc(const __bf16* __restrict__ A, const __bf16* __restrict__ B,
                   const float* __restrict__ kept, int eidx, int accumulate,
                   float* __restrict__ C) {
    const int lane = threadIdx.x & 31, w = threadIdx.x >> 5;
    const int wrow = w >> 2, wcol = w & 3;
    const int m0 = blockIdx.y * BM, n0 = blockIdx.x * BN;
    v8f acc[4][2];
    gemm_core_bf16<Fh>(A, B, m0, n0, acc);
    const int mo = (lane >> 4) * 8, ncol = lane & 15;
    #pragma unroll
    for (int i = 0; i < 4; ++i)
        #pragma unroll
        for (int j = 0; j < 2; ++j) {
            int gn = n0 + wcol * 32 + j * 16 + ncol;
            #pragma unroll
            for (int r = 0; r < 8; ++r) {
                int gm = m0 + wrow * 64 + i * 16 + mo + r;
                float kw = kept[(size_t)gm * En + eidx];
                float v  = kw * acc[i][j][r];
                float* p = &C[(size_t)gm * Dm + gn];
                *p = accumulate ? (*p + v) : v;
            }
        }
}

// ------------- GEMM3: logits = hn @ embb^T  (f32 out to d_out)
__global__ __launch_bounds__(256)
void moe_gemm3(const __bf16* __restrict__ A, const __bf16* __restrict__ B,
               float* __restrict__ C) {
    const int lane = threadIdx.x & 31, w = threadIdx.x >> 5;
    const int wrow = w >> 2, wcol = w & 3;
    const int m0 = blockIdx.y * BM, n0 = blockIdx.x * BN;
    v8f acc[4][2];
    gemm_core_bf16<Dm>(A, B, m0, n0, acc);
    const int mo = (lane >> 4) * 8, ncol = lane & 15;
    #pragma unroll
    for (int i = 0; i < 4; ++i)
        #pragma unroll
        for (int j = 0; j < 2; ++j) {
            int gn = n0 + wcol * 32 + j * 16 + ncol;
            #pragma unroll
            for (int r = 0; r < 8; ++r) {
                int gm = m0 + wrow * 64 + i * 16 + mo + r;
                C[(size_t)gm * VOC + gn] = acc[i][j][r];
            }
        }
}

// --------------------------- combine: h = (1-rho)*h + combined; refresh bf16
__global__ __launch_bounds__(256)
void moe_combine(float* __restrict__ h, __bf16* __restrict__ hbf,
                 const float* __restrict__ combined, const float* __restrict__ kept) {
    int t = blockIdx.x;
    float rho = 0.f;
    #pragma unroll
    for (int e = 0; e < En; ++e) rho += kept[(size_t)t * En + e];
    float om = 1.0f - rho;
    for (int d = threadIdx.x; d < Dm; d += 256) {
        size_t idx = (size_t)t * Dm + d;
        float v = om * h[idx] + combined[idx];
        h[idx]   = v;
        hbf[idx] = (__bf16)v;
    }
}

// ---------------------------------------------------------------- rmsnorm
__global__ __launch_bounds__(256)
void moe_rmsnorm(const float* __restrict__ h, const float* __restrict__ lnw,
                 __bf16* __restrict__ hn) {
    __shared__ float red[8];
    __shared__ float total;
    int t = blockIdx.x;
    float s = 0.f;
    for (int d = threadIdx.x; d < Dm; d += 256) {
        float x = h[(size_t)t * Dm + d];
        s += x * x;
    }
    #pragma unroll
    for (int off = 16; off > 0; off >>= 1) s += __shfl_down(s, off, 32);
    if ((threadIdx.x & 31) == 0) red[threadIdx.x >> 5] = s;
    __syncthreads();
    if (threadIdx.x == 0) {
        float tt = 0.f;
        #pragma unroll
        for (int i = 0; i < 8; ++i) tt += red[i];
        total = tt;
    }
    __syncthreads();
    float scale = rsqrtf(total / (float)Dm + 1e-6f);
    for (int d = threadIdx.x; d < Dm; d += 256) {
        float x = h[(size_t)t * Dm + d];
        hn[(size_t)t * Dm + d] = (__bf16)(x * scale * lnw[d]);
    }
}

// ---------------------------------------------------------------- launch
extern "C" void kernel_launch(void* const* d_in, const int* in_sizes, int n_in,
                              void* d_out, int out_size, void* d_ws, size_t ws_size,
                              hipStream_t stream) {
    const int*   ids = (const int*)d_in[0];
    const float* emb = (const float*)d_in[1];     // [VOC, Dm]
    const float* rw  = (const float*)d_in[2];     // [NHOP, Dm, En]
    const float* w1  = (const float*)d_in[3];     // [En, Dm, Fh]
    const float* w2  = (const float*)d_in[4];     // [En, Fh, Dm]
    const float* lnw = (const float*)d_in[5];     // [Dm]
    (void)in_sizes; (void)n_in; (void)out_size; (void)ws_size;

    char* ws = (char*)d_ws;
    const size_t OFF_H    = 0;                                   // f32  [Tn,Dm]
    const size_t OFF_HBF  = OFF_H    + (size_t)Tn * Dm * 4;      // bf16 [Tn,Dm]
    const size_t OFF_HN   = OFF_HBF  + (size_t)Tn * Dm * 2;      // bf16 [Tn,Dm]
    const size_t OFF_ACT  = OFF_HN   + (size_t)Tn * Dm * 2;      // bf16 [Tn,Fh]
    const size_t OFF_COMB = OFF_ACT  + (size_t)Tn * Fh * 2;      // f32  [Tn,Dm]
    const size_t OFF_PRB  = OFF_COMB + (size_t)Tn * Dm * 4;      // f32  [Tn,En]
    const size_t OFF_MSK  = OFF_PRB  + (size_t)Tn * En * 4;      // u32  [Tn]
    const size_t OFF_KPT  = OFF_MSK  + (size_t)Tn * 4;           // f32  [Tn,En]
    const size_t OFF_W1T  = OFF_KPT  + (size_t)Tn * En * 4;      // bf16 [En,Fh,Dm]
    const size_t OFF_W2T  = OFF_W1T  + (size_t)En * Dm * Fh * 2; // bf16 [En,Dm,Fh]
    const size_t OFF_EMBB = OFF_W2T  + (size_t)En * Fh * Dm * 2; // bf16 [VOC,Dm]

    float*    h     = (float*)(ws + OFF_H);
    __bf16*   hbf   = (__bf16*)(ws + OFF_HBF);
    __bf16*   hn    = (__bf16*)(ws + OFF_HN);
    __bf16*   act   = (__bf16*)(ws + OFF_ACT);
    float*    comb  = (float*)(ws + OFF_COMB);
    float*    probs = (float*)(ws + OFF_PRB);
    unsigned* maskb = (unsigned*)(ws + OFF_MSK);
    float*    kept  = (float*)(ws + OFF_KPT);
    __bf16*   w1t   = (__bf16*)(ws + OFF_W1T);
    __bf16*   w2t   = (__bf16*)(ws + OFF_W2T);
    __bf16*   embb  = (__bf16*)(ws + OFF_EMBB);

    conv_transpose_bf16<<<dim3(Fh / 32, Dm / 32, En), 256, 0, stream>>>(w1, w1t, Dm, Fh);
    conv_transpose_bf16<<<dim3(Dm / 32, Fh / 32, En), 256, 0, stream>>>(w2, w2t, Fh, Dm);
    conv_bf16<<<(size_t)VOC * Dm / 1024, 256, 0, stream>>>(emb, embb);

    moe_gather<<<Tn, 256, 0, stream>>>(ids, emb, h, hbf);

    for (int hop = 0; hop < NHOP; ++hop) {
        moe_router<<<Tn / 256, 256, 0, stream>>>(h, rw + (size_t)hop * Dm * En,
                                                 probs, maskb);
        moe_capacity<<<1, 32, 0, stream>>>(probs, maskb, kept);
        for (int e = 0; e < En; ++e) {
            moe_gemm1_gelu<<<dim3(Fh / BN, Tn / BM), 256, 0, stream>>>(
                hbf, w1t + (size_t)e * Dm * Fh, act);
            moe_gemm2_acc<<<dim3(Dm / BN, Tn / BM), 256, 0, stream>>>(
                act, w2t + (size_t)e * Fh * Dm, kept, e, /*accumulate=*/(e != 0), comb);
        }
        moe_combine<<<Tn, 256, 0, stream>>>(h, hbf, comb, kept);
    }

    moe_rmsnorm<<<Tn, 256, 0, stream>>>(h, lnw, hn);
    moe_gemm3<<<dim3(VOC / BN, Tn / BM), 256, 0, stream>>>(hn, embb, (float*)d_out);
}
